// AttnAgg_35553739276800
// MI455X (gfx1250) — matrix operand order repaired and checked by reading, not compiled, not run on hardware
//
#include <hip/hip_runtime.h>

#define MAIN_N  2048
#define OTHER_N 2048
#define QDIM    512
#define KDIM    512
#define MID     512
#define BATCH   32
#define NEG_BIG (-4294967296.0f)

typedef __attribute__((ext_vector_type(16))) __bf16 v16bf;
typedef __attribute__((ext_vector_type(8)))  __bf16 v8bf;
typedef __attribute__((ext_vector_type(4)))  __bf16 v4bf;
typedef __attribute__((ext_vector_type(8)))  float  v8f;
typedef __attribute__((ext_vector_type(4)))  unsigned int u32x4;
typedef __attribute__((ext_vector_type(4)))  int i32x4;
typedef __attribute__((ext_vector_type(8)))  int i32x8;

// ---------------------------------------------------------------------------
// WMMA helpers. CDNA5 bf16 WMMA: D(16x16,f32) = A(16x32,bf16) x B(32x16,bf16) + C
// Fragment lane layouts per ISA 7.12.2 (16-bit A 16x32 / B 32x16, wave32).
// ---------------------------------------------------------------------------
static __device__ inline v8f wmma_bf16(v16bf a, v16bf b, v8f c) {
  // (neg_a, A, neg_b, B, c_mod, C, reuse_a, reuse_b)
  return __builtin_amdgcn_wmma_f32_16x16x32_bf16(false, a, false, b, (short)0, c,
                                                 false, false);
}

// A fragment: row-major source, rows = M, contraction K contiguous.
// lanes 0-15: M=lane,    K in {k0..k0+7, k0+16..k0+23}
// lanes16-31: M=lane-16, K in {k0+8..k0+15, k0+24..k0+31}
static __device__ inline v16bf frag_a(const __bf16* base, int ld, int row0, int k0) {
  int lane = threadIdx.x & 31;
  int r  = row0 + (lane & 15);
  int kk = k0 + ((lane & 16) ? 8 : 0);
  const __bf16* p = base + r * ld + kk;
  v8bf lo = *(const v8bf*)p;          // K: kk .. kk+7
  v8bf hi = *(const v8bf*)(p + 16);   // K: kk+16 .. kk+23
  return __builtin_shufflevector(lo, hi, 0,1,2,3,4,5,6,7,8,9,10,11,12,13,14,15);
}

// B fragment for NT-GEMM: source is row-major [N rows x K cols] (i.e. B^T),
// so one lane reads two 8-element K runs of row N (same K order as frag_a).
// lanes 0-15: N=n0+lane,    K in {k0..k0+7, k0+16..k0+23}
// lanes16-31: N=n0+lane-16, K in {k0+8..k0+15, k0+24..k0+31}
static __device__ inline v16bf frag_b_nt(const __bf16* base, int ld, int n0, int k0) {
  int lane = threadIdx.x & 31;
  int n  = n0 + (lane & 15);
  int kk = k0 + ((lane & 16) ? 8 : 0);
  const __bf16* p = base + n * ld + kk;
  v8bf lo = *(const v8bf*)p;
  v8bf hi = *(const v8bf*)(p + 16);
  return __builtin_shufflevector(lo, hi, 0,1,2,3,4,5,6,7,8,9,10,11,12,13,14,15);
}

// ---------------------------------------------------------------------------
// TDM: scalar-issued 2D tile DMA global -> LDS (ISA ch.8 D# layout).
// 2-byte elements; tile_w elems per row, tile_h rows; dim0 stride in elems.
// ---------------------------------------------------------------------------
static __device__ inline void tdm_load_2d_bf16(unsigned lds_off, const void* gptr,
                                               unsigned tile_w, unsigned tile_h,
                                               unsigned stride_elems,
                                               unsigned dim0_elems) {
  unsigned long long ga = (unsigned long long)(size_t)gptr;
  u32x4 g0;
  g0[0] = 1u;                                          // count=1 (valid), user mode
  g0[1] = lds_off;                                     // lds_addr (bytes)
  g0[2] = (unsigned)(ga & 0xFFFFFFFFu);                // global_addr[31:0]
  g0[3] = (unsigned)((ga >> 32) & 0x01FFFFFFu)         // global_addr[56:32]
          | (2u << 30);                                // type = 2 ("image")
  const unsigned dim1 = 1u << 20;                      // large (no OOB clip on rows)
  i32x8 g1;
  g1[0] = (int)(1u << 16);                             // data_size=1 -> 2 bytes
  g1[1] = (int)((dim0_elems & 0xFFFFu) << 16);         // tensor_dim0[15:0]
  g1[2] = (int)(((dim0_elems >> 16) & 0xFFFFu)         // tensor_dim0[31:16]
                | ((dim1 & 0xFFFFu) << 16));           // tensor_dim1[15:0]
  g1[3] = (int)(((dim1 >> 16) & 0xFFFFu)               // tensor_dim1[31:16]
                | ((tile_w & 0xFFFFu) << 16));         // tile_dim0
  g1[4] = (int)(tile_h & 0xFFFFu);                     // tile_dim1 (tile_dim2=0)
  g1[5] = (int)stride_elems;                           // tensor_dim0_stride[31:0]
  g1[6] = 0;                                           // stride[47:32]=0, dim1_stride lo=0
  g1[7] = 0;
  i32x4 z4 = {0, 0, 0, 0};
#if __clang_major__ >= 23
  i32x8 z8 = {0, 0, 0, 0, 0, 0, 0, 0};
  __builtin_amdgcn_tensor_load_to_lds(g0, g1, z4, z4, z8, 0);
#else
  __builtin_amdgcn_tensor_load_to_lds(g0, g1, z4, z4, 0);
#endif
}

// ---------------------------------------------------------------------------
// f32 -> bf16 conversion (vectorized x4)
// ---------------------------------------------------------------------------
__global__ __launch_bounds__(256)
void cvt_f32_bf16(const float* __restrict__ s, __bf16* __restrict__ d, int n) {
  int i = (blockIdx.x * blockDim.x + threadIdx.x) * 4;
  if (i + 3 < n) {
    float4 v = *(const float4*)(s + i);
    v4bf o = {(__bf16)v.x, (__bf16)v.y, (__bf16)v.z, (__bf16)v.w};
    *(v4bf*)(d + i) = o;
  }
}

// ---------------------------------------------------------------------------
// Linear: C[m,n] = sum_k A[m,k] * W[n,k] + bias[n]  (NT-GEMM, bf16 in/out)
// Block = 256 thr (8 waves), block tile 32(m) x 256(n); wave tile 16x64.
// ---------------------------------------------------------------------------
__global__ __launch_bounds__(256)
void linear_nt_kernel(const __bf16* __restrict__ A, const __bf16* __restrict__ W,
                      const float* __restrict__ bias, __bf16* __restrict__ C,
                      int M, int N, int Kd) {
  int wave = threadIdx.x >> 5, lane = threadIdx.x & 31;
  int wm = wave >> 2;          // 0..1
  int wn = wave & 3;           // 0..3
  int m0 = blockIdx.y * 32 + wm * 16;
  int n0 = blockIdx.x * 256 + wn * 64;

  v8f zero = {0.f,0.f,0.f,0.f,0.f,0.f,0.f,0.f};
  v8f acc[4] = {zero, zero, zero, zero};

  for (int k0 = 0; k0 < Kd; k0 += 32) {
    __builtin_prefetch(A + m0 * Kd + k0 + 128, 0, 1);
    v16bf a = frag_a(A, Kd, m0, k0);
#pragma unroll
    for (int j = 0; j < 4; ++j) {
      v16bf b = frag_b_nt(W, Kd, n0 + j * 16, k0);
      acc[j] = wmma_bf16(a, b, acc[j]);
    }
  }

  int cn = lane & 15;
  int rb = (lane & 16) ? 8 : 0;
#pragma unroll
  for (int j = 0; j < 4; ++j) {
    int n = n0 + j * 16 + cn;
    float bv = bias[n];
#pragma unroll
    for (int r = 0; r < 8; ++r) {
      int m = m0 + rb + r;
      C[m * N + n] = (__bf16)(acc[j][r] + bv);
    }
  }
}

// ---------------------------------------------------------------------------
// Attention logits: S[m,o] = (Q[m,:] . K[o,:]) / sqrt(MID), masked.
// NT-GEMM, wave tile 16(m) x 64(o), block tile 32 x 256.
// ---------------------------------------------------------------------------
__global__ __launch_bounds__(256)
void attn_kernel(const __bf16* __restrict__ Q, const __bf16* __restrict__ Kb,
                 const unsigned char* __restrict__ mask, float* __restrict__ S) {
  int wave = threadIdx.x >> 5, lane = threadIdx.x & 31;
  int wm = wave >> 2, wn = wave & 3;
  int m0 = blockIdx.y * 32 + wm * 16;
  int o0 = blockIdx.x * 256 + wn * 64;

  v8f zero = {0.f,0.f,0.f,0.f,0.f,0.f,0.f,0.f};
  v8f acc[4] = {zero, zero, zero, zero};

  for (int k0 = 0; k0 < MID; k0 += 32) {
    __builtin_prefetch(Q + m0 * MID + k0 + 128, 0, 1);
    v16bf a = frag_a(Q, MID, m0, k0);
#pragma unroll
    for (int j = 0; j < 4; ++j) {
      v16bf b = frag_b_nt(Kb, MID, o0 + j * 16, k0);
      acc[j] = wmma_bf16(a, b, acc[j]);
    }
  }

  const float scale = 0.044194173824159216f;  // 1/sqrt(512)
  int cn = lane & 15;
  int rb = (lane & 16) ? 8 : 0;
#pragma unroll
  for (int j = 0; j < 4; ++j) {
#pragma unroll
    for (int r = 0; r < 8; ++r) {
      int m = m0 + rb + r;
      int o = o0 + j * 16 + cn;
      float sv = acc[j][r] * scale;
      S[m * OTHER_N + o] = mask[m * OTHER_N + o] ? NEG_BIG : sv;
    }
  }
}

// ---------------------------------------------------------------------------
// Row softmax over 2048 cols; writes bf16 probabilities.
// ---------------------------------------------------------------------------
__global__ __launch_bounds__(256)
void softmax_kernel(const float* __restrict__ S, __bf16* __restrict__ P) {
  __shared__ float red[256];
  int row = blockIdx.x, t = threadIdx.x;
  const float* sr = S + (size_t)row * OTHER_N;

  float v[8];
  float mx = -3.402823466e38f;
#pragma unroll
  for (int i = 0; i < 8; ++i) { v[i] = sr[t + i * 256]; mx = fmaxf(mx, v[i]); }
  red[t] = mx; __syncthreads();
  for (int s = 128; s > 0; s >>= 1) {
    if (t < s) red[t] = fmaxf(red[t], red[t + s]);
    __syncthreads();
  }
  mx = red[0]; __syncthreads();

  float sum = 0.f;
#pragma unroll
  for (int i = 0; i < 8; ++i) { v[i] = __expf(v[i] - mx); sum += v[i]; }
  red[t] = sum; __syncthreads();
  for (int s = 128; s > 0; s >>= 1) {
    if (t < s) red[t] += red[t + s];
    __syncthreads();
  }
  float inv = 1.0f / red[0];

  __bf16* pr = P + (size_t)row * OTHER_N;
#pragma unroll
  for (int i = 0; i < 8; ++i) pr[t + i * 256] = (__bf16)(v[i] * inv);
}

// ---------------------------------------------------------------------------
// Aggregation: out[b,m,k] = sum_o P[m,o] * fix[b,o] * other[o,k]
// Grid: (k-tiles=4, m-tiles=16, b=32). Block 256 thr = 8 waves.
// Block tile 128(m) x 128(k); o-loop in steps of 32 staged through LDS.
//  - P tile (128x32, natural layout) staged by the Tensor Data Mover,
//    issued by wave 0, synchronized with s_wait_tensorcnt.
//  - other tile (32x128) staged transposed with fix[b,o] scale folded in.
// Wave tile 32(m) x 64(k) = 2x4 accumulators, A-fragment reuse.
// ---------------------------------------------------------------------------
__global__ __launch_bounds__(256)
void agg_kernel(const __bf16* __restrict__ P, const __bf16* __restrict__ other_bf,
                const float* __restrict__ fix, float* __restrict__ out) {
  __shared__ __bf16 lds_p[128 * 32];   // P tile  [m][o] row-major (A operand)
  __shared__ __bf16 lds_oT[128 * 32];  // other^T [k][o] (B^T row-major, NT form)

  int k0 = blockIdx.x * 128;   // 0..511 within batch slice
  int m0 = blockIdx.y * 128;
  int b  = blockIdx.z;
  int tid = threadIdx.x;
  int wave = tid >> 5, lane = tid & 31;
  int wm = wave >> 1;          // 0..3 -> m sub-tile (32 rows)
  int wk = wave & 1;           // 0..1 -> k sub-tile (64 cols)

  const float* fixb = fix + (size_t)b * OTHER_N;
  unsigned lds_p_off = (unsigned)(size_t)&lds_p[0];

  v8f zero = {0.f,0.f,0.f,0.f,0.f,0.f,0.f,0.f};
  v8f acc[2][4];
#pragma unroll
  for (int i = 0; i < 2; ++i)
#pragma unroll
    for (int j = 0; j < 4; ++j) acc[i][j] = zero;

  for (int o0 = 0; o0 < OTHER_N; o0 += 32) {
    // Stage P tile via TDM: 128 rows x 32 bf16, row stride 2048 elems.
    if (wave == 0) {
      tdm_load_2d_bf16(lds_p_off, P + (size_t)m0 * OTHER_N + o0,
                       /*tile_w=*/32, /*tile_h=*/128,
                       /*stride=*/OTHER_N, /*dim0=*/OTHER_N);
    }
    // Stage other tile scaled by fix[b,o], transposed to [k][o].
    // 32(o) x 128(k) = 512 x 16B chunks, 2 chunks per thread.
#pragma unroll
    for (int c = tid; c < 512; c += 256) {
      int o  = c >> 4;           // 0..31
      int kc = (c & 15) * 8;     // 0,8,...,120
      float fs = fixb[o0 + o];
      v8bf vv = *(const v8bf*)(other_bf + (size_t)(o0 + o) * KDIM + k0 + kc);
#pragma unroll
      for (int e = 0; e < 8; ++e)
        lds_oT[(kc + e) * 32 + o] = (__bf16)((float)vv[e] * fs);
    }
    if (o0 + 32 < OTHER_N)
      __builtin_prefetch(other_bf + (size_t)(o0 + 32) * KDIM + k0, 0, 1);
    if (wave == 0) __builtin_amdgcn_s_wait_tensorcnt(0);
    __syncthreads();

    v16bf a0 = frag_a(lds_p, 32, wm * 32, 0);
    v16bf a1 = frag_a(lds_p, 32, wm * 32 + 16, 0);
#pragma unroll
    for (int j = 0; j < 4; ++j) {
      v16bf bf = frag_b_nt(lds_oT, 32, wk * 64 + j * 16, 0);
      acc[0][j] = wmma_bf16(a0, bf, acc[0][j]);
      acc[1][j] = wmma_bf16(a1, bf, acc[1][j]);
    }
    __syncthreads();
  }

  int cn = lane & 15;
  int rb = (lane & 16) ? 8 : 0;
  float* outb = out + (size_t)b * MAIN_N * KDIM;
#pragma unroll
  for (int i = 0; i < 2; ++i)
#pragma unroll
    for (int j = 0; j < 4; ++j)
#pragma unroll
      for (int r = 0; r < 8; ++r) {
        int m = m0 + wm * 32 + i * 16 + rb + r;
        int k = k0 + wk * 64 + j * 16 + cn;
        outb[(size_t)m * KDIM + k] = acc[i][j][r];
      }
}

// ---------------------------------------------------------------------------
extern "C" void kernel_launch(void* const* d_in, const int* in_sizes, int n_in,
                              void* d_out, int out_size, void* d_ws, size_t ws_size,
                              hipStream_t stream) {
  (void)in_sizes; (void)n_in; (void)out_size; (void)ws_size;

  const float*         main_feat  = (const float*)d_in[0];
  const float*         other_feat = (const float*)d_in[1];
  const float*         fix        = (const float*)d_in[2];
  const unsigned char* mask       = (const unsigned char*)d_in[3];
  const float*         Wq         = (const float*)d_in[4];
  const float*         bq         = (const float*)d_in[5];
  const float*         Wk         = (const float*)d_in[6];
  const float*         bk         = (const float*)d_in[7];
  float*               out        = (float*)d_out;

  char* ws = (char*)d_ws;
  size_t off = 0;
  auto alloc = [&](size_t bytes) {
    char* p = ws + off;
    off += (bytes + 255) & ~(size_t)255;
    return p;
  };
  __bf16* main_bf  = (__bf16*)alloc((size_t)MAIN_N * QDIM * 2);
  __bf16* other_bf = (__bf16*)alloc((size_t)OTHER_N * KDIM * 2);
  __bf16* wq_bf    = (__bf16*)alloc((size_t)MID * QDIM * 2);
  __bf16* wk_bf    = (__bf16*)alloc((size_t)MID * KDIM * 2);
  __bf16* q_bf     = (__bf16*)alloc((size_t)MAIN_N * MID * 2);
  __bf16* k_bf     = (__bf16*)alloc((size_t)OTHER_N * MID * 2);
  float*  S        = (float*)alloc((size_t)MAIN_N * OTHER_N * 4);
  __bf16* P        = (__bf16*)alloc((size_t)MAIN_N * OTHER_N * 2);

  // 1) f32 -> bf16 conversions
  {
    int n1 = MAIN_N * QDIM;
    int n2 = OTHER_N * KDIM;
    int n3 = MID * QDIM;
    int n4 = MID * KDIM;
    cvt_f32_bf16<<<dim3((n1 / 4 + 255) / 256), 256, 0, stream>>>(main_feat,  main_bf,  n1);
    cvt_f32_bf16<<<dim3((n2 / 4 + 255) / 256), 256, 0, stream>>>(other_feat, other_bf, n2);
    cvt_f32_bf16<<<dim3((n3 / 4 + 255) / 256), 256, 0, stream>>>(Wq, wq_bf, n3);
    cvt_f32_bf16<<<dim3((n4 / 4 + 255) / 256), 256, 0, stream>>>(Wk, wk_bf, n4);
  }

  // 2) Q = main @ Wq^T + bq ; K = other @ Wk^T + bk  (bf16 WMMA)
  linear_nt_kernel<<<dim3(MID / 256, MAIN_N / 32), 256, 0, stream>>>(
      main_bf, wq_bf, bq, q_bf, MAIN_N, MID, QDIM);
  linear_nt_kernel<<<dim3(MID / 256, OTHER_N / 32), 256, 0, stream>>>(
      other_bf, wk_bf, bk, k_bf, OTHER_N, MID, KDIM);

  // 3) S = mask(Q K^T / sqrt(MID))
  attn_kernel<<<dim3(OTHER_N / 256, MAIN_N / 32), 256, 0, stream>>>(q_bf, k_bf, mask, S);

  // 4) P = softmax_rows(S) as bf16
  softmax_kernel<<<dim3(MAIN_N), 256, 0, stream>>>(S, P);

  // 5) out[b] = P @ (diag(fix[b]) other)  (bf16 WMMA + TDM staging)
  agg_kernel<<<dim3(KDIM / 128, MAIN_N / 128, BATCH), 256, 0, stream>>>(
      P, other_bf, fix, out);
}
